// DiVeQDetach_78426102825289
// MI455X (gfx1250) — matrix operand order repaired and checked by reading, not compiled, not run on hardware
//
#include <hip/hip_runtime.h>

typedef __attribute__((ext_vector_type(2)))  float  v2f;
typedef __attribute__((ext_vector_type(4)))  float  v4f;
typedef __attribute__((ext_vector_type(8)))  float  v8f;
typedef __attribute__((ext_vector_type(4)))  __bf16 v4bf;
typedef __attribute__((ext_vector_type(16))) __bf16 v16bf;

#define NUM_EMB   1024
#define DIM       64
#define CHUNK     128                 // codes per LDS chunk
#define NCHUNK    (NUM_EMB / CHUNK)   // 8
#define CBS       72                  // bf16 per code row: 144B pitch, 16B aligned, conflict-free
#define WAVES     8
#define ROWS_PER_WAVE 16
#define ROWS_PER_BLOCK (WAVES * ROWS_PER_WAVE)  // 128
#define N_TOTAL   32768

__global__ __launch_bounds__(256) void vq_wmma_kernel(
    const float* __restrict__ z, const float* __restrict__ cb,
    float* __restrict__ zq_out, float* __restrict__ idx_out,
    float* __restrict__ block_partial)
{
  __shared__ __bf16 lds_hi[CHUNK * CBS];        // 18432 B
  __shared__ __bf16 lds_lo[CHUNK * CBS];        // 18432 B
  __shared__ float  lds_csq[NUM_EMB];           // 4096 B
  __shared__ int    lds_idx[ROWS_PER_BLOCK];    // 512 B
  __shared__ float  lds_red[WAVES];             // 32 B

  const int tid  = threadIdx.x;
  const int wave = tid >> 5;
  const int lane = tid & 31;
  const int col  = lane & 15;          // N (codes) / M (z-rows) lane position
  const int g    = lane >> 4;          // lane half selects K sub-blocks
  const int rows_base = blockIdx.x * ROWS_PER_BLOCK + wave * ROWS_PER_WAVE;

  // ---- c_sq for all 1024 codes, exact f32 (once) ----
  for (int c = tid; c < NUM_EMB; c += 256) {
    const v4f* row = reinterpret_cast<const v4f*>(cb + c * DIM);
    float s = 0.f;
    #pragma unroll
    for (int q = 0; q < DIM / 4; ++q) {
      v4f v = row[q];
      s += v.x * v.x + v.y * v.y + v.z * v.z + v.w * v.w;
    }
    lds_csq[c] = s;
  }

  // ---- resident A fragments: split z into bf16 hi/lo, 2 K-chunks of 32 ----
  // A k order used here (contiguous, not AMD's table order): row = lane&15; elem j   <-> K = g*16 + j (j=0..7)
  //                                       elem 8+j <-> K = g*16 + 8 + j
  v16bf a_hi[2], a_lo[2];
  {
    const float* zr = z + (rows_base + col) * DIM;
    #pragma unroll
    for (int c = 0; c < 2; ++c) {
      #pragma unroll
      for (int j = 0; j < 8; ++j) {
        float x0 = zr[32 * c + g * 16 + j];
        float x1 = zr[32 * c + g * 16 + 8 + j];
        __bf16 h0 = (__bf16)x0;
        __bf16 h1 = (__bf16)x1;
        a_hi[c][j]     = h0;  a_lo[c][j]     = (__bf16)(x0 - (float)h0);
        a_hi[c][8 + j] = h1;  a_lo[c][8 + j] = (__bf16)(x1 - (float)h1);
      }
    }
  }

  float sbest[8];
  int   ibest[8];
  #pragma unroll
  for (int v = 0; v < 8; ++v) { sbest[v] = 3.4e38f; ibest[v] = 0; }

  for (int ch = 0; ch < NCHUNK; ++ch) {
    __syncthreads();  // csq done (first iter) / previous chunk consumed
    // stage 128 codes: f32 -> bf16 hi/lo planes. 2048 float4 by 256 threads.
    const float* src = cb + ch * CHUNK * DIM;
    #pragma unroll
    for (int it = 0; it < 8; ++it) {
      int e  = (it * 256 + tid) * 4;
      int cl = e >> 6;        // code within chunk
      int kk = e & 63;        // k offset
      v4f val = *reinterpret_cast<const v4f*>(src + e);
      v4bf hi, lo;
      #pragma unroll
      for (int q = 0; q < 4; ++q) {
        __bf16 h = (__bf16)val[q];
        hi[q] = h;
        lo[q] = (__bf16)(val[q] - (float)h);
      }
      *reinterpret_cast<v4bf*>(&lds_hi[cl * CBS + kk]) = hi;
      *reinterpret_cast<v4bf*>(&lds_lo[cl * CBS + kk]) = lo;
    }
    __syncthreads();

    // 8 tiles of 16 codes; fully unrolled -> many independent WMMA chains.
    // B k order used here (contiguous, same as A): col = lane&15; elem j <-> K = g*16 + j.
    #pragma unroll
    for (int mt = 0; mt < CHUNK / 16; ++mt) {
      const int m_base = ch * CHUNK + mt * 16;
      const int cbase  = (mt * 16 + col) * CBS + g * 16;

      v16bf bh0 = *reinterpret_cast<const v16bf*>(&lds_hi[cbase]);       // K 0..31
      v16bf bh1 = *reinterpret_cast<const v16bf*>(&lds_hi[cbase + 32]);  // K 32..63
      v16bf bl0 = *reinterpret_cast<const v16bf*>(&lds_lo[cbase]);
      v16bf bl1 = *reinterpret_cast<const v16bf*>(&lds_lo[cbase + 32]);

      // dots ~= zhi*chi + zhi*clo + zlo*chi  (f32 accumulate, ~2^-17 rel err)
      // Two independent 3-WMMA chains -> intra-tile ILP, shorter serial depth.
      v8f acc0 = {0.f, 0.f, 0.f, 0.f, 0.f, 0.f, 0.f, 0.f};
      v8f acc1 = {0.f, 0.f, 0.f, 0.f, 0.f, 0.f, 0.f, 0.f};
      acc0 = __builtin_amdgcn_wmma_f32_16x16x32_bf16(false, a_hi[0], false, bh0, (short)0, acc0, false, false);
      acc1 = __builtin_amdgcn_wmma_f32_16x16x32_bf16(false, a_hi[1], false, bh1, (short)0, acc1, false, false);
      acc0 = __builtin_amdgcn_wmma_f32_16x16x32_bf16(false, a_hi[0], false, bl0, (short)0, acc0, false, false);
      acc1 = __builtin_amdgcn_wmma_f32_16x16x32_bf16(false, a_hi[1], false, bl1, (short)0, acc1, false, false);
      acc0 = __builtin_amdgcn_wmma_f32_16x16x32_bf16(false, a_lo[0], false, bh0, (short)0, acc0, false, false);
      acc1 = __builtin_amdgcn_wmma_f32_16x16x32_bf16(false, a_lo[1], false, bh1, (short)0, acc1, false, false);

      const float csq = lds_csq[m_base + col];
      const int   id  = m_base + col;
      // Per-lane candidate ids are strictly ascending across tiles, so a
      // strict '<' already gives first-occurrence (lowest index) semantics.
      #pragma unroll
      for (int v = 0; v < 8; ++v) {
        float sc = csq - 2.0f * (acc0[v] + acc1[v]);
        if (sc < sbest[v]) { sbest[v] = sc; ibest[v] = id; }
      }
    }
  }

  // ---- argmin reduce across the 16 lanes that share a row ----
  #pragma unroll
  for (int v = 0; v < 8; ++v) {
    float s = sbest[v]; int i = ibest[v];
    #pragma unroll
    for (int m = 1; m < 16; m <<= 1) {
      float so = __shfl_xor(s, m, 32);
      int   io = __shfl_xor(i, m, 32);
      if (so < s || (so == s && io < i)) { s = so; i = io; }
    }
    sbest[v] = s; ibest[v] = i;
  }
  if (col == 0) {
    #pragma unroll
    for (int v = 0; v < 8; ++v) {
      int row = v + (g << 3);                     // 0..7 lower half, 8..15 upper
      lds_idx[wave * 16 + row] = ibest[v];
      idx_out[rows_base + row] = (float)ibest[v]; // indices as f32 in flat tuple
    }
  }
  __syncthreads();

  // ---- stage 2: z_q, loss (exact f32 from global). 2 cols/lane/row. ----
  float lsum = 0.f;
  #pragma unroll 1
  for (int r = 0; r < 16; ++r) {
    const int grow = rows_base + r;
    const int bi   = lds_idx[wave * 16 + r];
    v2f zv = *reinterpret_cast<const v2f*>(z  + grow * DIM + lane * 2);
    v2f cv = *reinterpret_cast<const v2f*>(cb + bi   * DIM + lane * 2);
    float dx = cv.x - zv.x, dy = cv.y - zv.y;
    float d2 = dx * dx + dy * dy;
    lsum += d2;
    float rs = d2;
    #pragma unroll
    for (int m = 1; m < 32; m <<= 1) rs += __shfl_xor(rs, m, 32);
    float mag = sqrtf(rs);
    float sca = mag / (mag + 1e-8f);               // magnitude * d/(magnitude+eps)
    v2f o; o.x = zv.x + sca * dx; o.y = zv.y + sca * dy;
    *reinterpret_cast<v2f*>(zq_out + grow * DIM + lane * 2) = o;
  }
  #pragma unroll
  for (int m = 1; m < 32; m <<= 1) lsum += __shfl_xor(lsum, m, 32);
  if (lane == 0) lds_red[wave] = lsum;
  __syncthreads();
  if (tid == 0) {
    float bs = 0.f;
    #pragma unroll
    for (int w = 0; w < WAVES; ++w) bs += lds_red[w];
    block_partial[blockIdx.x] = bs;   // deterministic two-pass loss
  }
}

__global__ __launch_bounds__(256) void loss_reduce_kernel(
    const float* __restrict__ partial, float* __restrict__ loss)
{
  __shared__ float sm[256];
  int tid = threadIdx.x;
  sm[tid] = partial[tid];              // exactly 256 blocks of partials
  __syncthreads();
  #pragma unroll
  for (int s = 128; s > 0; s >>= 1) {
    if (tid < s) sm[tid] += sm[tid + s];
    __syncthreads();
  }
  if (tid == 0) {
    // loss = codebook_loss + 0.25*commitment_loss = 1.25 * mean(d^2)
    loss[0] = sm[0] * (1.25f / (float)(N_TOTAL * DIM));
  }
}

extern "C" void kernel_launch(void* const* d_in, const int* in_sizes, int n_in,
                              void* d_out, int out_size, void* d_ws, size_t ws_size,
                              hipStream_t stream) {
  const float* z  = (const float*)d_in[0];   // [32768, 64]
  const float* cb = (const float*)d_in[1];   // [1024, 64]
  float* out  = (float*)d_out;
  float* zq   = out;                         // 2097152 floats
  float* loss = out + (size_t)N_TOTAL * DIM; // 1 float
  float* idxo = loss + 1;                    // 32768 floats
  float* part = (float*)d_ws;                // 256 floats scratch

  vq_wmma_kernel<<<N_TOTAL / ROWS_PER_BLOCK, 256, 0, stream>>>(z, cb, zq, idxo, part);
  loss_reduce_kernel<<<1, 256, 0, stream>>>(part, loss);
}
